// ForwardPass_29137058136067
// MI455X (gfx1250) — matrix operand order repaired and checked by reading, not compiled, not run on hardware
//
#include <hip/hip_runtime.h>
#include <hip/hip_bf16.h>
#include <math.h>

typedef __attribute__((ext_vector_type(16))) __bf16 v16bf;
typedef __attribute__((ext_vector_type(8)))  float  v8f;
typedef __attribute__((ext_vector_type(4)))  unsigned int v4u;
typedef __attribute__((ext_vector_type(8)))  int    v8i;
typedef __attribute__((ext_vector_type(4)))  int    v4i;

#define B_   2
#define N_   2048
#define DM_  1024
#define H_   16
#define DK_  64
#define BLK_ 128
#define TB_  16
#define NEG_ (-1e30f)

__device__ __forceinline__ float red_max16(float x) {
  x = fmaxf(x, __shfl_xor(x, 1, 32));
  x = fmaxf(x, __shfl_xor(x, 2, 32));
  x = fmaxf(x, __shfl_xor(x, 4, 32));
  x = fmaxf(x, __shfl_xor(x, 8, 32));
  return x;
}
__device__ __forceinline__ float red_sum16(float x) {
  x += __shfl_xor(x, 1, 32);
  x += __shfl_xor(x, 2, 32);
  x += __shfl_xor(x, 4, 32);
  x += __shfl_xor(x, 8, 32);
  return x;
}

__global__ __launch_bounds__(256, 1) void bs_flash_fwd(
    const float* __restrict__ q, const float* __restrict__ k,
    const float* __restrict__ v, const float* __restrict__ taup,
    const int* __restrict__ mask, const int* __restrict__ bsm,
    float* __restrict__ outO, float* __restrict__ outL, float* __restrict__ outM)
{
  __shared__ __bf16 Ks[BLK_][DK_ + 2];   // [key][dk], +2 pad -> 33-dword stride
  __shared__ __bf16 Vt[DK_][BLK_ + 2];   // [dk][key] transposed, +2 pad
  __shared__ __bf16 Psb[8][16][34];      // per-wave P (bf16) transpose scratch
  __shared__ int    Ms[BLK_][BLK_];      // token-mask tile (TDM destination, contiguous)

  const int qb = blockIdx.x, h = blockIdx.y, b = blockIdx.z;
  const int tid  = threadIdx.x;
  const int lane = tid & 31, wave = tid >> 5;
  const int ln   = lane & 15, half = lane >> 4;

  const float scale = taup[0] * 0.125f;     // tau / sqrt(d_k=64)
  const int qblk0 = qb * BLK_;
  const int q0 = qblk0 + wave * 16;         // this wave's 16 q-rows

  // ---- Q tile (16 rows x 64 dk) as two bf16 A operands (16x32 each)
  v16bf aQ0, aQ1;
  {
    const float* qrow = q + ((size_t)(b * N_ + q0 + ln) * DM_ + h * DK_);
    #pragma unroll
    for (int r = 0; r < 8; ++r) {
      int kk = (r >> 2) * 16 + half * 8 + (r & 3) * 2;   // A-layout K index
      float2 f0 = *(const float2*)(qrow + kk);
      float2 f1 = *(const float2*)(qrow + 32 + kk);
      aQ0[2*r] = (__bf16)f0.x;  aQ0[2*r+1] = (__bf16)f0.y;
      aQ1[2*r] = (__bf16)f1.x;  aQ1[2*r+1] = (__bf16)f1.y;
    }
  }

  v8f O0 = {0,0,0,0,0,0,0,0}, O1 = O0, O2 = O0, O3 = O0;
  float m_run[8], l_run[8];
  #pragma unroll
  for (int r = 0; r < 8; ++r) { m_run[r] = -__builtin_inff(); l_run[r] = 0.f; }

  // staging role for this thread (K/V convert-stage)
  const int kr  = tid >> 1;
  const int dk0 = (tid & 1) * 32;

  for (int j = 0; j < TB_; ++j) {
    if (bsm[qb * TB_ + j] == 0) continue;     // block-sparse skip (uniform)

    __syncthreads();   // previous block's LDS consumers are done

    // ---- issue TDM: mask tile (128 q-rows x 128 keys, int32) -> LDS, async
    if (wave == 0) {
      unsigned int lds_off = (unsigned int)(uintptr_t)(&Ms[0][0]);
      unsigned long long ga =
          (unsigned long long)(uintptr_t)(mask + (size_t)b * N_ * N_ +
                                          (size_t)qblk0 * N_ + (size_t)j * BLK_);
      v4u g0;
      g0.x = 1u;                                     // count=1, user mode
      g0.y = lds_off;                                // lds_addr
      g0.z = (unsigned int)(ga & 0xFFFFFFFFu);       // global_addr[31:0]
      g0.w = (unsigned int)((ga >> 32) & 0x01FFFFFFu) | (2u << 30);  // addr[56:32], type=2
      v8i g1;
      g1[0] = (int)(2u << 16);            // workgroup_mask=0, data_size=2 (4B)
      g1[1] = (int)(2048u << 16);         // tensor_dim0[15:0]=2048 at bits[31:16]
      g1[2] = (int)(2048u << 16);         // tensor_dim0 hi=0 ; tensor_dim1[15:0]=2048
      g1[3] = (int)(128u << 16);          // tensor_dim1 hi=0 ; tile_dim0=128
      g1[4] = (int)(128u);                // tile_dim1=128 ; tile_dim2=0
      g1[5] = (int)(2048u);               // tensor_dim0_stride[31:0]=2048
      g1[6] = 0;                          // stride0 hi ; tensor_dim1_stride lo16
      g1[7] = 0;
      v4i gz4  = {0, 0, 0, 0};
      v8i gz8  = {0, 0, 0, 0, 0, 0, 0, 0};
      __builtin_amdgcn_tensor_load_to_lds(g0, g1, gz4, gz4, gz8, 0);
    }

    // ---- prefetch next allowed block's K/V rows while this block is staged
    {
      int jn = -1;
      for (int t = j + 1; t < TB_; ++t)
        if (bsm[qb * TB_ + t] != 0) { jn = t; break; }
      if (jn >= 0) {
        const float* kpn = k + ((size_t)(b * N_ + jn * BLK_ + kr) * DM_ + h * DK_ + dk0);
        const float* vpn = v + ((size_t)(b * N_ + jn * BLK_ + kr) * DM_ + h * DK_ + dk0);
        __builtin_prefetch(kpn, 0, 1);
        __builtin_prefetch(vpn, 0, 1);
      }
    }

    // ---- cooperative convert-stage: K row-major bf16, V transposed bf16
    {
      const float* kp = k + ((size_t)(b * N_ + j * BLK_ + kr) * DM_ + h * DK_ + dk0);
      const float* vp = v + ((size_t)(b * N_ + j * BLK_ + kr) * DM_ + h * DK_ + dk0);
      #pragma unroll
      for (int i = 0; i < 32; i += 2) {
        float2 fk = *(const float2*)(kp + i);
        Ks[kr][dk0 + i]     = (__bf16)fk.x;
        Ks[kr][dk0 + i + 1] = (__bf16)fk.y;
        float2 fv = *(const float2*)(vp + i);
        Vt[dk0 + i][kr]     = (__bf16)fv.x;
        Vt[dk0 + i + 1][kr] = (__bf16)fv.y;
      }
    }

    if (wave == 0) __builtin_amdgcn_s_wait_tensorcnt(0);   // mask tile landed
    __syncthreads();

    for (int kb = 0; kb < 4; ++kb) {          // 32 keys per chunk
      const int kb32 = kb * 32;

      // ---- S = Q @ K^T for two 16-key subtiles
      v8f s_t[2];
      #pragma unroll
      for (int s = 0; s < 2; ++s) {
        v16bf bk0, bk1;
        const int keyr = kb32 + s * 16 + ln;  // B-operand: lane = N = key
        #pragma unroll
        for (int r = 0; r < 8; ++r) {
          int d0 = (r >> 2) * 16 + half * 8 + (r & 3) * 2;   // B-layout K index (= dk)
          bk0[2*r]   = Ks[keyr][d0];
          bk0[2*r+1] = Ks[keyr][d0 + 1];
          bk1[2*r]   = Ks[keyr][32 + d0];
          bk1[2*r+1] = Ks[keyr][32 + d0 + 1];
        }
        v8f zero = {0,0,0,0,0,0,0,0};
        v8f t = __builtin_amdgcn_wmma_f32_16x16x32_bf16(false, aQ0, false, bk0,
                                                        (short)0, zero, false, false);
        s_t[s] = __builtin_amdgcn_wmma_f32_16x16x32_bf16(false, aQ1, false, bk1,
                                                         (short)0, t, false, false);
      }

      // ---- mask (from LDS) + scale + online softmax (fp32 statistics)
      float p0[8], p1[8], alpha[8];
      #pragma unroll
      for (int r = 0; r < 8; ++r) {
        const int lqr = wave * 16 + r + half * 8;   // row within the 128-row tile
        int mv0 = Ms[lqr][kb32 + ln];
        int mv1 = Ms[lqr][kb32 + 16 + ln];
        float sm0 = (mv0 != 0) ? s_t[0][r] * scale : NEG_;
        float sm1 = (mv1 != 0) ? s_t[1][r] * scale : NEG_;
        float rmax = red_max16(fmaxf(sm0, sm1));
        float newm = fmaxf(m_run[r], rmax);
        alpha[r] = (m_run[r] == newm) ? 1.f : __expf(m_run[r] - newm);
        p0[r] = (sm0 > -1e29f) ? __expf(sm0 - newm) : 0.f;
        p1[r] = (sm1 > -1e29f) ? __expf(sm1 - newm) : 0.f;
        float rsum = red_sum16(p0[r] + p1[r]);
        l_run[r] = l_run[r] * alpha[r] + rsum;
        m_run[r] = newm;
      }
      #pragma unroll
      for (int r = 0; r < 8; ++r) {
        O0[r] *= alpha[r]; O1[r] *= alpha[r]; O2[r] *= alpha[r]; O3[r] *= alpha[r];
      }

      // ---- transpose P (C/D layout -> A layout) via per-wave LDS scratch (bf16)
      #pragma unroll
      for (int r = 0; r < 8; ++r) {
        Psb[wave][r + half * 8][ln]      = (__bf16)p0[r];
        Psb[wave][r + half * 8][16 + ln] = (__bf16)p1[r];
      }
      asm volatile("" ::: "memory");      // wave-private region; HW LDS in-order per wave
      v16bf aP;
      #pragma unroll
      for (int r = 0; r < 8; ++r) {
        int kk = (r >> 2) * 16 + half * 8 + (r & 3) * 2;
        aP[2*r]   = Psb[wave][ln][kk];
        aP[2*r+1] = Psb[wave][ln][kk + 1];
      }
      asm volatile("" ::: "memory");

      // ---- O(16x64) += P(16x32) @ V(32x64), four 16-dk column tiles
      #pragma unroll
      for (int t = 0; t < 4; ++t) {
        v16bf bv;
        const int dk = t * 16 + ln;           // B-operand: lane = N = dk column
        #pragma unroll
        for (int r = 0; r < 8; ++r) {
          int kkey = kb32 + (r >> 2) * 16 + half * 8 + (r & 3) * 2;  // K = key; pair contiguous in Vt row
          bv[2*r]   = Vt[dk][kkey];
          bv[2*r+1] = Vt[dk][kkey + 1];
        }
        v8f& Ot = (t == 0) ? O0 : (t == 1) ? O1 : (t == 2) ? O2 : O3;
        Ot = __builtin_amdgcn_wmma_f32_16x16x32_bf16(false, aP, false, bv,
                                                     (short)0, Ot, false, false);
      }
    }
  }

  // ---- epilogue: O /= l (0 if l==0), write O, l, m
  float inv[8];
  #pragma unroll
  for (int r = 0; r < 8; ++r) inv[r] = (l_run[r] > 0.f) ? (1.f / l_run[r]) : 0.f;

  #pragma unroll
  for (int t = 0; t < 4; ++t) {
    const v8f& Ot = (t == 0) ? O0 : (t == 1) ? O1 : (t == 2) ? O2 : O3;
    #pragma unroll
    for (int r = 0; r < 8; ++r) {
      const int qr = q0 + r + half * 8;
      outO[(size_t)(b * N_ + qr) * DM_ + h * DK_ + t * 16 + ln] = Ot[r] * inv[r];
    }
  }
  if (ln == 0) {   // lane 0 -> rows 0..7, lane 16 -> rows 8..15
    #pragma unroll
    for (int r = 0; r < 8; ++r) {
      const int qr = q0 + r + half * 8;
      size_t idx = (size_t)(b * H_ + h) * N_ + qr;
      outL[idx] = l_run[r];
      outM[idx] = (l_run[r] > 0.f) ? m_run[r] : -__builtin_inff();
    }
  }
}

extern "C" void kernel_launch(void* const* d_in, const int* in_sizes, int n_in,
                              void* d_out, int out_size, void* d_ws, size_t ws_size,
                              hipStream_t stream) {
  const float* q    = (const float*)d_in[0];
  const float* k    = (const float*)d_in[1];
  const float* v    = (const float*)d_in[2];
  const float* tau  = (const float*)d_in[3];
  const int*   mask = (const int*)d_in[4];
  const int*   bsm  = (const int*)d_in[5];

  float* out  = (float*)d_out;
  float* outO = out;                                   // (B,N,D_MODEL)
  float* outL = out + (size_t)B_ * N_ * DM_;           // (B,H,N,1)
  float* outM = outL + (size_t)B_ * H_ * N_;           // (B,H,N,1)

  dim3 grid(TB_, H_, B_);   // (q-block, head, batch)
  bs_flash_fwd<<<grid, 256, 0, stream>>>(q, k, v, tau, mask, bsm, outO, outL, outM);
}
